// StandardAttention_2808908611943
// MI455X (gfx1250) — matrix operand order repaired and checked by reading, not compiled, not run on hardware
//
#include <hip/hip_runtime.h>
#include <hip/hip_bf16.h>

// ---------------------------------------------------------------------------
// MHA forward for MI455X (gfx1250): bf16 WMMA everywhere, f32 accumulate.
//   x[4,2048,1024] fp32 -> qkv proj -> attention (flash, transposed scores)
//   -> out proj -> out fp32.
// Round 3: pre-convert x / w_qkv / w_out to bf16 once, so both GEMM inner
// loops are pure bf16 b128 loads feeding v_wmma (no per-iteration cvt chains,
// half the load bytes).
// ---------------------------------------------------------------------------

typedef __attribute__((ext_vector_type(16))) __bf16 v16bf;
typedef __attribute__((ext_vector_type(8)))  __bf16 v8bf;
typedef __attribute__((ext_vector_type(4)))  __bf16 v4bf;
typedef __attribute__((ext_vector_type(8)))  float  v8f;
typedef __attribute__((ext_vector_type(4)))  float  v4f;

#define D_MODEL 1024
#define NHEAD   16
#define HD      64
#define BATCH   4
#define SEQ     2048
#define M_TOTAL (BATCH * SEQ)        // 8192 rows
#define VT_STRIDE 40                 // LDS row stride (elems): 80B rows, 16B-aligned segs

static __device__ __forceinline__ __bf16 f2bf(float x) { return (__bf16)x; }

static __device__ __forceinline__ v8f wmma_bf16(v16bf a, v16bf b, v8f c) {
    // D = A(16x32) * B(32x16) + C, f32 accum
    return __builtin_amdgcn_wmma_f32_16x16x32_bf16(
        /*neg_a=*/false, a, /*neg_b=*/false, b,
        /*c_mod=*/(short)0, c, /*reuse_a=*/false, /*reuse_b=*/false);
}

static __device__ __forceinline__ v16bf join8(v8bf lo, v8bf hi) {
    v16bf r;
#pragma unroll
    for (int i = 0; i < 8; ++i) { r[i] = lo[i]; r[8 + i] = hi[i]; }
    return r;
}

// A-fragment (16x32 bf16): lane = row, elems = 16 contiguous K starting at
// k=16h (matches B-fragment). Two 16B loads.
static __device__ __forceinline__ v16bf afrag_bf16(const __bf16* __restrict__ row, int hh) {
    return join8(*(const v8bf*)(row + 16 * hh),
                 *(const v8bf*)(row + 16 * hh + 8));
}

// B-fragment (32x16 bf16): lane = col, elems = 16 contiguous K starting at
// k=16h (matches A-fragment). Two 16B loads.
static __device__ __forceinline__ v16bf bfrag_bf16(const __bf16* __restrict__ row) {
    return join8(*(const v8bf*)(row), *(const v8bf*)(row + 8));
}

// ---------------------------------------------------------------------------
// Kernel 0: elementwise fp32 -> bf16 conversion (vectorized by 4).
// ---------------------------------------------------------------------------
__global__ __launch_bounds__(256) void cvt_f32_bf16_kernel(
    const float* __restrict__ src, __bf16* __restrict__ dst, int n4)
{
    const int i = blockIdx.x * blockDim.x + threadIdx.x;
    if (i < n4) {
        v4f a = *(const v4f*)(src + 4 * (size_t)i);
        v4bf r;
#pragma unroll
        for (int j = 0; j < 4; ++j) r[j] = f2bf(a[j]);
        *(v4bf*)(dst + 4 * (size_t)i) = r;
    }
}

// ---------------------------------------------------------------------------
// Kernel 1: QKV projection. y = xb @ wb^T + b_qkv, split into bf16 q/k/v
// workspaces laid out [B, H, T, HD]. One wave -> 32x64 C tile
// (2 M-tiles x 4 N-tiles; each B-fragment feeds two WMMAs).
// ---------------------------------------------------------------------------
__global__ __launch_bounds__(256) void qkv_gemm_kernel(
    const __bf16* __restrict__ xb, const __bf16* __restrict__ wb,
    const float* __restrict__ bias,
    __bf16* __restrict__ qo, __bf16* __restrict__ ko, __bf16* __restrict__ vo)
{
    const int lane = threadIdx.x & 31;
    const int wave = threadIdx.x >> 5;
    const int lm = lane & 15;        // row/col within tile
    const int hh = lane >> 4;        // lane half

    const int NCOLG = (3 * D_MODEL) / 64;          // 48 col-groups
    const int job = blockIdx.x * 8 + wave;         // 12288 wave-jobs
    const int mt = job / NCOLG;
    const int ng = job % NCOLG;
    const int m0 = mt * 32;
    const int n0 = ng * 64;

    v8f acc[2][4];
#pragma unroll
    for (int mi = 0; mi < 2; ++mi)
#pragma unroll
        for (int nt = 0; nt < 4; ++nt)
#pragma unroll
            for (int j = 0; j < 8; ++j) acc[mi][nt][j] = 0.0f;

    const __bf16* arow0 = xb + (size_t)(m0 + lm) * D_MODEL;
    const __bf16* arow1 = xb + (size_t)(m0 + 16 + lm) * D_MODEL;

    for (int k0 = 0; k0 < D_MODEL; k0 += 32) {
        v16bf a0 = afrag_bf16(arow0 + k0, hh);
        v16bf a1 = afrag_bf16(arow1 + k0, hh);
#pragma unroll
        for (int nt = 0; nt < 4; ++nt) {
            const int n = n0 + nt * 16 + lm;
            v16bf b = bfrag_bf16(wb + (size_t)n * D_MODEL + k0 + 16 * hh);
            acc[0][nt] = wmma_bf16(a0, b, acc[0][nt]);
            acc[1][nt] = wmma_bf16(a1, b, acc[1][nt]);
        }
    }

    // Store: decompose n -> (q/k/v, head, d), m -> (batch, t); bf16 output.
#pragma unroll
    for (int nt = 0; nt < 4; ++nt) {
        const int n = n0 + nt * 16 + lm;
        const float bv = bias[n];
        const int sel  = n / D_MODEL;
        const int rem  = n % D_MODEL;
        const int head = rem / HD;
        const int d    = rem % HD;
        __bf16* dst = (sel == 0) ? qo : ((sel == 1) ? ko : vo);
#pragma unroll
        for (int mi = 0; mi < 2; ++mi) {
#pragma unroll
            for (int j = 0; j < 8; ++j) {
                const int mm = m0 + mi * 16 + j + 8 * hh;    // C/D row
                const int bb = mm / SEQ;
                const int tt = mm % SEQ;
                const size_t addr = (((size_t)bb * NHEAD + head) * SEQ + tt) * HD + d;
                dst[addr] = f2bf(acc[mi][nt][j] + bv);
            }
        }
    }
}

// ---------------------------------------------------------------------------
// Kernel 2: attention. One wave owns a 32-query tile of one (b,head) and
// streams all 2048 keys in 32-key chunks with online softmax.
// Scores computed transposed: S^T = K * Q^T (per-lane softmax state; the
// query index lives in the lane, so max/sum merge is one shfl_xor(16)).
// Output accumulated transposed: O^T += V^T * P^T  (V^T staged in LDS).
// Every K / V^T A-fragment is reused by both query-column tiles.
// ---------------------------------------------------------------------------
__global__ __launch_bounds__(256) void attn_kernel(
    const __bf16* __restrict__ q, const __bf16* __restrict__ k,
    const __bf16* __restrict__ v, __bf16* __restrict__ attn_out)
{
    __shared__ __bf16 lds_vt[8][HD * VT_STRIDE];   // per-wave V^T tile (5120B)

    const int lane = threadIdx.x & 31;
    const int wave = threadIdx.x >> 5;
    const int lm = lane & 15;
    const int hh = lane >> 4;
    __bf16* vt = lds_vt[wave];

    const int job = blockIdx.x * 8 + wave;     // 4096 jobs = 64 bh * 64 qtiles
    const int qt32 = job & 63;                 // 32-query tile index
    const int bh   = job >> 6;
    const int q0 = qt32 * 32;
    const int b    = bh >> 4;
    const int head = bh & 15;

    const size_t base = (size_t)bh * SEQ * HD;
    const __bf16* qp = q + base;
    const __bf16* kp = k + base;
    const __bf16* vp = v + base;

    // Q^T B-fragments, loaded once: lane col = query, elems = head-dim 8h+e, 16+8h+e.
    v16bf bq[2][2];                            // [qtile][hd-half]
#pragma unroll
    for (int qt = 0; qt < 2; ++qt) {
        const __bf16* qrow = qp + (size_t)(q0 + qt * 16 + lm) * HD + 8 * hh;
        bq[qt][0] = join8(*(const v8bf*)(qrow),      *(const v8bf*)(qrow + 16));
        bq[qt][1] = join8(*(const v8bf*)(qrow + 32), *(const v8bf*)(qrow + 48));
    }

    const float scale = 0.125f;                 // 1/sqrt(64)
    float run_m[2] = { -__builtin_inff(), -__builtin_inff() };
    float run_s[2] = { 0.0f, 0.0f };
    v8f ot[2][4];
#pragma unroll
    for (int qt = 0; qt < 2; ++qt)
#pragma unroll
        for (int dt = 0; dt < 4; ++dt)
#pragma unroll
            for (int j = 0; j < 8; ++j) ot[qt][dt][j] = 0.0f;

    for (int c = 0; c < SEQ; c += 32) {
        // ---- stage V^T chunk: lane loads V row (c+lane), scatters to LDS ----
        {
            const __bf16* vrow = vp + (size_t)(c + lane) * HD;
            v8bf r[8];
#pragma unroll
            for (int i = 0; i < 8; ++i) r[i] = *(const v8bf*)(vrow + 8 * i);
#pragma unroll
            for (int i = 0; i < 8; ++i)
#pragma unroll
                for (int e = 0; e < 8; ++e)
                    vt[(i * 8 + e) * VT_STRIDE + lane] = r[i][e];
        }

        // ---- scores: S^T tiles [qtile][keytile]; K A-frags reused ---------
        v8f st[2][2];
#pragma unroll
        for (int kt = 0; kt < 2; ++kt) {
            const __bf16* krow = kp + (size_t)(c + kt * 16 + lm) * HD;
            v16bf a0 = join8(*(const v8bf*)(krow + 8 * hh),
                             *(const v8bf*)(krow + 16 + 8 * hh));
            v16bf a1 = join8(*(const v8bf*)(krow + 32 + 8 * hh),
                             *(const v8bf*)(krow + 48 + 8 * hh));
#pragma unroll
            for (int qt = 0; qt < 2; ++qt) {
                v8f s;
#pragma unroll
                for (int j = 0; j < 8; ++j) s[j] = 0.0f;
                s = wmma_bf16(a0, bq[qt][0], s);
                s = wmma_bf16(a1, bq[qt][1], s);
                st[qt][kt] = s;
            }
        }

        // ---- online softmax + P^T B-fragment per query tile ----------------
        v16bf pb[2];
#pragma unroll
        for (int qt = 0; qt < 2; ++qt) {
            float cm = -__builtin_inff();
#pragma unroll
            for (int j = 0; j < 8; ++j)
                cm = fmaxf(cm, fmaxf(st[qt][0][j], st[qt][1][j]));
            cm = fmaxf(cm, __shfl_xor(cm, 16));
            const float nm = fmaxf(run_m[qt], cm);
            const float alpha = __expf(scale * (run_m[qt] - nm));
            run_m[qt] = nm;

            float p0[8], p1[8];
            float ls = 0.0f;
#pragma unroll
            for (int j = 0; j < 8; ++j) {
                p0[j] = __expf(scale * (st[qt][0][j] - nm));
                p1[j] = __expf(scale * (st[qt][1][j] - nm));
                ls += p0[j] + p1[j];
            }
            ls += __shfl_xor(ls, 16);
            run_s[qt] = run_s[qt] * alpha + ls;

#pragma unroll
            for (int dt = 0; dt < 4; ++dt)
#pragma unroll
                for (int j = 0; j < 8; ++j) ot[qt][dt][j] *= alpha;

            // lane half 0 needs keys 0..15, half 1 needs keys 16..31; swap
            // the missing 8 values with the partner lane (xor 16).
#pragma unroll
            for (int j = 0; j < 8; ++j) {
                const float r0 = __shfl_xor(p0[j], 16);
                const float r1 = __shfl_xor(p1[j], 16);
                const float e_lo = hh ? r1 : p0[j];      // elems 0..7
                const float e_hi = hh ? p1[j] : r0;      // elems 8..15
                pb[qt][j]     = f2bf(e_lo);
                pb[qt][8 + j] = f2bf(e_hi);
            }
        }

        // ---- O^T += V^T * P^T : 4 d-tiles; V^T A-frags reused by both qt ---
#pragma unroll
        for (int dt = 0; dt < 4; ++dt) {
            const __bf16* lrow = vt + (dt * 16 + lm) * VT_STRIDE;
            v16bf a = join8(*(const v8bf*)(lrow + 16 * hh),
                            *(const v8bf*)(lrow + 16 * hh + 8));
#pragma unroll
            for (int qt = 0; qt < 2; ++qt)
                ot[qt][dt] = wmma_bf16(a, pb[qt], ot[qt][dt]);
        }
    }

    // ---- finalize: divide by row sum, store bf16 into [B,T,D_MODEL] --------
#pragma unroll
    for (int qt = 0; qt < 2; ++qt) {
        const float inv = 1.0f / run_s[qt];
        const int qq = q0 + qt * 16 + lm;
#pragma unroll
        for (int dt = 0; dt < 4; ++dt) {
#pragma unroll
            for (int j = 0; j < 8; ++j) {
                const int d = dt * 16 + j + 8 * hh;      // D row = j + 8*half
                const size_t addr =
                    ((size_t)b * SEQ + qq) * D_MODEL + head * HD + d;
                attn_out[addr] = f2bf(ot[qt][dt][j] * inv);
            }
        }
    }
}

// ---------------------------------------------------------------------------
// Kernel 3: output projection. out = attn(bf16) @ w_out^T + b_out (fp32 out).
// One wave -> 32x64 C tile (2 M-tiles x 4 N-tiles).
// ---------------------------------------------------------------------------
__global__ __launch_bounds__(256) void out_proj_kernel(
    const __bf16* __restrict__ a_in, const __bf16* __restrict__ wb,
    const float* __restrict__ bias, float* __restrict__ out)
{
    const int lane = threadIdx.x & 31;
    const int wave = threadIdx.x >> 5;
    const int lm = lane & 15;
    const int hh = lane >> 4;

    const int NCOLG = D_MODEL / 64;                // 16 col-groups
    const int job = blockIdx.x * 8 + wave;         // 4096 wave-jobs
    const int mt = job / NCOLG;
    const int ng = job % NCOLG;
    const int m0 = mt * 32;
    const int n0 = ng * 64;

    v8f acc[2][4];
#pragma unroll
    for (int mi = 0; mi < 2; ++mi)
#pragma unroll
        for (int nt = 0; nt < 4; ++nt)
#pragma unroll
            for (int j = 0; j < 8; ++j) acc[mi][nt][j] = 0.0f;

    const __bf16* arow0 = a_in + (size_t)(m0 + lm) * D_MODEL;
    const __bf16* arow1 = a_in + (size_t)(m0 + 16 + lm) * D_MODEL;

    for (int k0 = 0; k0 < D_MODEL; k0 += 32) {
        v16bf a0 = afrag_bf16(arow0 + k0, hh);
        v16bf a1 = afrag_bf16(arow1 + k0, hh);
#pragma unroll
        for (int nt = 0; nt < 4; ++nt) {
            const int n = n0 + nt * 16 + lm;
            v16bf b = bfrag_bf16(wb + (size_t)n * D_MODEL + k0 + 16 * hh);
            acc[0][nt] = wmma_bf16(a0, b, acc[0][nt]);
            acc[1][nt] = wmma_bf16(a1, b, acc[1][nt]);
        }
    }

#pragma unroll
    for (int nt = 0; nt < 4; ++nt) {
        const int n = n0 + nt * 16 + lm;
        const float bv = bias[n];
#pragma unroll
        for (int mi = 0; mi < 2; ++mi) {
#pragma unroll
            for (int j = 0; j < 8; ++j) {
                const int mm = m0 + mi * 16 + j + 8 * hh;
                out[(size_t)mm * D_MODEL + n] = acc[mi][nt][j] + bv;
            }
        }
    }
}

// ---------------------------------------------------------------------------
extern "C" void kernel_launch(void* const* d_in, const int* in_sizes, int n_in,
                              void* d_out, int out_size, void* d_ws, size_t ws_size,
                              hipStream_t stream)
{
    const float* x     = (const float*)d_in[0];
    const float* w_qkv = (const float*)d_in[1];
    const float* b_qkv = (const float*)d_in[2];
    const float* w_out = (const float*)d_in[3];
    const float* b_out = (const float*)d_in[4];
    float* out = (float*)d_out;

    // Workspace layout (bf16): q, k, v each [B,H,T,HD]; attn [B,T,D];
    // then bf16 copies of x, w_qkv, w_out.
    const size_t n_elems   = (size_t)M_TOTAL * D_MODEL;      // 8.39M
    const size_t wqkv_elems = (size_t)3 * D_MODEL * D_MODEL; // 3.15M
    const size_t wout_elems = (size_t)D_MODEL * D_MODEL;     // 1.05M
    __bf16* qb = (__bf16*)d_ws;
    __bf16* kb = qb + n_elems;
    __bf16* vb = kb + n_elems;
    __bf16* ab = vb + n_elems;
    __bf16* xb = ab + n_elems;
    __bf16* wqkvb = xb + n_elems;
    __bf16* woutb = wqkvb + wqkv_elems;

    // Pre-convert fp32 operands to bf16 (each element converted exactly once).
    cvt_f32_bf16_kernel<<<(int)(n_elems / 4 + 255) / 256, 256, 0, stream>>>(x, xb, (int)(n_elems / 4));
    cvt_f32_bf16_kernel<<<(int)(wqkv_elems / 4 + 255) / 256, 256, 0, stream>>>(w_qkv, wqkvb, (int)(wqkv_elems / 4));
    cvt_f32_bf16_kernel<<<(int)(wout_elems / 4 + 255) / 256, 256, 0, stream>>>(w_out, woutb, (int)(wout_elems / 4));

    // QKV: (8192/32)*(3072/64) = 12288 wave-jobs / 8 waves per block
    qkv_gemm_kernel<<<1536, 256, 0, stream>>>(xb, wqkvb, b_qkv, qb, kb, vb);
    // Attention: 64 (b,h) * 64 32-query tiles = 4096 wave-jobs / 8
    attn_kernel<<<512, 256, 0, stream>>>(qb, kb, vb, ab);
    // Out projection: (8192/32)*(1024/64) = 4096 wave-jobs / 8
    out_proj_kernel<<<512, 256, 0, stream>>>(ab, w_out ? woutb : woutb, b_out, out);
}
